// EncoderBlock_33913061769384
// MI455X (gfx1250) — matrix operand order repaired and checked by reading, not compiled, not run on hardware
//
#include <hip/hip_runtime.h>
#include <stdint.h>

#define D_MODEL 1024
#define SEQ     2048
#define BATCH   4
#define NH      16
#define HD      64
#define M_TOTAL (BATCH * SEQ)   // 8192 rows

typedef __attribute__((ext_vector_type(16))) __bf16 v16bf;
typedef __attribute__((ext_vector_type(8)))  float  v8f;

union Frag16 { uint32_t u[8]; v16bf v; };

#if defined(__has_builtin)
# if __has_builtin(__builtin_amdgcn_global_load_async_to_lds_b128)
#  define HAVE_ASYNC_LDS 1
# endif
#endif
#ifndef HAVE_ASYNC_LDS
# define HAVE_ASYNC_LDS 0
#endif

#if HAVE_ASYNC_LDS
typedef int v4i_gcc __attribute__((vector_size(16)));
typedef __attribute__((address_space(1))) v4i_gcc glb_v4i;
typedef __attribute__((address_space(3))) v4i_gcc lds_v4i;
__device__ __forceinline__ void async_copy16(const uint16_t* g, uint16_t* l) {
  // 16 bytes per lane, memory -> LDS, tracked by ASYNCcnt
  __builtin_amdgcn_global_load_async_to_lds_b128((glb_v4i*)g, (lds_v4i*)l, 0, 0);
}
#endif

__device__ __forceinline__ void wait_async_lds() {
#if HAVE_ASYNC_LDS
# if __has_builtin(__builtin_amdgcn_s_wait_asynccnt)
  __builtin_amdgcn_s_wait_asynccnt(0);
# else
  asm volatile("s_wait_asynccnt 0x0" ::: "memory");
# endif
#endif
}

__device__ __forceinline__ uint16_t f2bf(float f) {
  uint32_t x = __float_as_uint(f);
  return (uint16_t)((x + 0x7FFFu + ((x >> 16) & 1u)) >> 16);
}

// ---------------------------------------------------------------- cast f32->bf16
__global__ void cast_bf16_kernel(const float* __restrict__ in,
                                 uint16_t* __restrict__ out, int n) {
  int i = blockIdx.x * blockDim.x + threadIdx.x;
  int stride = gridDim.x * blockDim.x;
  for (; i < n; i += stride) out[i] = f2bf(in[i]);
}

// ---------------------------------------------------------------- GEMM: Y = A (MxK) * W^T (W is NxK)
// 256 threads = 8 waves; tile 128(M) x 128(N); K-step 64; double-buffered LDS.
template <bool BIAS, bool RELU, bool WF32, bool WBF16>
__global__ __launch_bounds__(256)
void gemm_bt_kernel(const uint16_t* __restrict__ A, const uint16_t* __restrict__ W,
                    const float* __restrict__ bias,
                    float* __restrict__ outF, uint16_t* __restrict__ outB,
                    int M, int N, int K) {
  __shared__ uint16_t As[2][128 * 64];
  __shared__ uint16_t Bs[2][128 * 64];

  const int nTilesN = N >> 7;
  const int m0 = (blockIdx.x / nTilesN) << 7;
  const int n0 = (blockIdx.x % nTilesN) << 7;
  const uint16_t* gA = A + (size_t)m0 * K;
  const uint16_t* gW = W + (size_t)n0 * K;

  const int t    = threadIdx.x;
  const int wave = t >> 5;
  const int lane = t & 31;
  const int hi   = lane >> 4;
  const int ln   = lane & 15;
  const int wm   = wave >> 1;   // rows wm*32
  const int wn   = wave & 1;    // cols wn*64

  // staging map: 4 x 16B chunks per array per thread (128x64 bf16 tile)
  int crow[4], ccol[4];
#pragma unroll
  for (int p = 0; p < 4; ++p) {
    int chunk = t + p * 256;
    crow[p] = chunk >> 3;
    ccol[p] = (chunk & 7) * 8;
  }

  v8f acc[2][4] = {};

#if HAVE_ASYNC_LDS
  auto fetch = [&](int kk0, int buf) {
#pragma unroll
    for (int p = 0; p < 4; ++p) {
      async_copy16(&gA[(size_t)crow[p] * K + kk0 + ccol[p]], &As[buf][crow[p] * 64 + ccol[p]]);
      async_copy16(&gW[(size_t)crow[p] * K + kk0 + ccol[p]], &Bs[buf][crow[p] * 64 + ccol[p]]);
    }
  };
  auto commit = [&](int) {};
#else
  uint4 ra[4], rb[4];
  auto fetch = [&](int kk0, int) {
#pragma unroll
    for (int p = 0; p < 4; ++p) {
      ra[p] = *(const uint4*)&gA[(size_t)crow[p] * K + kk0 + ccol[p]];
      rb[p] = *(const uint4*)&gW[(size_t)crow[p] * K + kk0 + ccol[p]];
    }
  };
  auto commit = [&](int buf) {
#pragma unroll
    for (int p = 0; p < 4; ++p) {
      *(uint4*)&As[buf][crow[p] * 64 + ccol[p]] = ra[p];
      *(uint4*)&Bs[buf][crow[p] * 64 + ccol[p]] = rb[p];
    }
  };
#endif

  const int NS = K >> 6;     // 64-wide K stages
  fetch(0, 0);
  commit(0);
  wait_async_lds();
  __syncthreads();

  for (int s = 0; s < NS; ++s) {
    const int cur = s & 1;
    if (s + 1 < NS) fetch((s + 1) << 6, cur ^ 1);

#pragma unroll
    for (int ks = 0; ks < 2; ++ks) {
      Frag16 af[2];
#pragma unroll
      for (int tm = 0; tm < 2; ++tm) {
        const uint32_t* row =
            (const uint32_t*)&As[cur][(wm * 32 + tm * 16 + ln) * 64 + ks * 32];
#pragma unroll
        for (int v = 0; v < 4; ++v) {
          af[tm].u[v]     = row[hi * 4 + v];       // K = hi*8 + 2v
          af[tm].u[4 + v] = row[8 + hi * 4 + v];   // K = 16 + hi*8 + 2v
        }
      }
#pragma unroll
      for (int tn = 0; tn < 4; ++tn) {
        Frag16 bf;
        const uint32_t* row =
            (const uint32_t*)&Bs[cur][(wn * 64 + tn * 16 + ln) * 64 + ks * 32];
#pragma unroll
        for (int v = 0; v < 4; ++v) {
          bf.u[v]     = row[hi * 4 + v];       // K = hi*8 + 2v
          bf.u[4 + v] = row[8 + hi * 4 + v];   // K = 16 + hi*8 + 2v
        }
#pragma unroll
        for (int tm = 0; tm < 2; ++tm)
          acc[tm][tn] = __builtin_amdgcn_wmma_f32_16x16x32_bf16(
              false, af[tm].v, false, bf.v, (short)0, acc[tm][tn], false, false);
      }
    }

    if (s + 1 < NS) commit(cur ^ 1);
    wait_async_lds();
    __syncthreads();
  }

#pragma unroll
  for (int tm = 0; tm < 2; ++tm)
#pragma unroll
    for (int tn = 0; tn < 4; ++tn)
#pragma unroll
      for (int r = 0; r < 8; ++r) {
        int gm = m0 + wm * 32 + tm * 16 + hi * 8 + r;
        int gn = n0 + wn * 64 + tn * 16 + ln;
        float val = acc[tm][tn][r];
        if (BIAS) val += bias[gn];
        if (RELU) val = fmaxf(val, 0.0f);
        size_t idx = (size_t)gm * N + gn;
        if (WF32)  outF[idx] = val;
        if (WBF16) outB[idx] = f2bf(val);
      }
}

// ---------------------------------------------------------------- flash attention
// grid = B * NH * (SEQ/128); block = 256 (8 waves, 16 q-rows each)
__global__ __launch_bounds__(256)
void flash_attn_kernel(const uint16_t* __restrict__ q, const uint16_t* __restrict__ k,
                       const uint16_t* __restrict__ v, float* __restrict__ out) {
  __shared__ uint16_t Ks[64 * 64];        // row-major [kk][d]
  __shared__ uint16_t Vs[64 * 64];        // transposed [d][kk]
  __shared__ uint16_t Ps[8 * 16 * 64];    // per-wave P tiles [wave][m][kk]

  const int bx = blockIdx.x;
  const int qt = bx & 15;
  const int h  = (bx >> 4) & 15;
  const int b  = bx >> 8;

  const int t    = threadIdx.x;
  const int wave = t >> 5;
  const int lane = t & 31;
  const int hi   = lane >> 4;
  const int ln   = lane & 15;
  const int qrow0 = qt * 128 + wave * 16;

  // Q fragments stay resident (A-layout, 2 K-steps of 32 over hd=64)
  Frag16 qf[2];
#pragma unroll
  for (int ks = 0; ks < 2; ++ks) {
    const uint32_t* row = (const uint32_t*)
        &q[((size_t)(b * SEQ + qrow0 + ln)) * D_MODEL + h * HD + ks * 32];
#pragma unroll
    for (int vv = 0; vv < 4; ++vv) {
      qf[ks].u[vv]     = row[hi * 4 + vv];
      qf[ks].u[4 + vv] = row[8 + hi * 4 + vv];
    }
  }

  v8f o[4] = {};
  float mrun[8], lrun[8];
#pragma unroll
  for (int r = 0; r < 8; ++r) { mrun[r] = -1e30f; lrun[r] = 0.0f; }

  const int rowL = t >> 3;          // 0..31, 8 threads per 64-elem row
  const int colL = (t & 7) * 8;

  for (int it = 0; it < SEQ / 64; ++it) {
    const int kk0 = it * 64;
    __syncthreads();

#if HAVE_ASYNC_LDS
#pragma unroll
    for (int p = 0; p < 2; ++p) {
      int chunk = t + p * 256;
      int rr = chunk >> 3;
      int cc = (chunk & 7) * 8;
      async_copy16(&k[((size_t)(b * SEQ + kk0 + rr)) * D_MODEL + h * HD + cc],
                   &Ks[rr * 64 + cc]);
    }
#else
#pragma unroll
    for (int p = 0; p < 2; ++p) {
      int rr = rowL + p * 32;
      *(uint4*)&Ks[rr * 64 + colL] =
          *(const uint4*)&k[((size_t)(b * SEQ + kk0 + rr)) * D_MODEL + h * HD + colL];
    }
#endif
    // V tile transposed into LDS (element scatter; stays on sync path)
#pragma unroll
    for (int p = 0; p < 2; ++p) {
      int rr = rowL + p * 32;
      uint4 vv4 = *(const uint4*)&v[((size_t)(b * SEQ + kk0 + rr)) * D_MODEL + h * HD + colL];
      const uint16_t* pv = (const uint16_t*)&vv4;
#pragma unroll
      for (int e = 0; e < 8; ++e) Vs[(colL + e) * 64 + rr] = pv[e];
    }
    wait_async_lds();
    __syncthreads();

    // scores S = Q * K^T  (16 x 64)
    v8f s[4] = {};
#pragma unroll
    for (int nt = 0; nt < 4; ++nt) {
      const int kkl = nt * 16 + ln;
#pragma unroll
      for (int ks = 0; ks < 2; ++ks) {
        Frag16 bfr;
        const uint32_t* row = (const uint32_t*)&Ks[kkl * 64 + ks * 32];
#pragma unroll
        for (int vv = 0; vv < 4; ++vv) {
          bfr.u[vv]     = row[hi * 4 + vv];
          bfr.u[4 + vv] = row[8 + hi * 4 + vv];
        }
        s[nt] = __builtin_amdgcn_wmma_f32_16x16x32_bf16(
            false, qf[ks].v, false, bfr.v, (short)0, s[nt], false, false);
      }
    }

    // online softmax (scale = 1/sqrt(64) = 0.125)
    const float sc = 0.125f;
#pragma unroll
    for (int r = 0; r < 8; ++r) {
      float mx = -1e30f;
#pragma unroll
      for (int nt = 0; nt < 4; ++nt) mx = fmaxf(mx, s[nt][r]);
#pragma unroll
      for (int msk = 1; msk < 16; msk <<= 1) mx = fmaxf(mx, __shfl_xor(mx, msk, 16));
      mx *= sc;
      float mnew  = fmaxf(mrun[r], mx);
      float scale = __expf(mrun[r] - mnew);
      mrun[r] = mnew;
      float sum = 0.0f;
#pragma unroll
      for (int nt = 0; nt < 4; ++nt) {
        float p = __expf(s[nt][r] * sc - mnew);
        s[nt][r] = p;
        sum += p;
      }
#pragma unroll
      for (int msk = 1; msk < 16; msk <<= 1) sum += __shfl_xor(sum, msk, 16);
      lrun[r] = lrun[r] * scale + sum;
#pragma unroll
      for (int nt = 0; nt < 4; ++nt) o[nt][r] *= scale;
    }

    // C-layout P -> A-layout via per-wave LDS tile
#pragma unroll
    for (int nt = 0; nt < 4; ++nt)
#pragma unroll
      for (int r = 0; r < 8; ++r)
        Ps[wave * 1024 + (hi * 8 + r) * 64 + nt * 16 + ln] = f2bf(s[nt][r]);
    __syncthreads();

    // O += P * V
#pragma unroll
    for (int ks = 0; ks < 2; ++ks) {
      Frag16 pf;
      const uint32_t* prow = (const uint32_t*)&Ps[wave * 1024 + ln * 64 + ks * 32];
#pragma unroll
      for (int vv = 0; vv < 4; ++vv) {
        pf.u[vv]     = prow[hi * 4 + vv];
        pf.u[4 + vv] = prow[8 + hi * 4 + vv];
      }
#pragma unroll
      for (int dt = 0; dt < 4; ++dt) {
        Frag16 vf;
        const uint32_t* vrow = (const uint32_t*)&Vs[(dt * 16 + ln) * 64 + ks * 32];
#pragma unroll
        for (int vv = 0; vv < 4; ++vv) {
          vf.u[vv]     = vrow[hi * 4 + vv];
          vf.u[4 + vv] = vrow[8 + hi * 4 + vv];
        }
        o[dt] = __builtin_amdgcn_wmma_f32_16x16x32_bf16(
            false, pf.v, false, vf.v, (short)0, o[dt], false, false);
      }
    }
  }

#pragma unroll
  for (int dt = 0; dt < 4; ++dt)
#pragma unroll
    for (int r = 0; r < 8; ++r) {
      int gq = qrow0 + hi * 8 + r;
      int gd = h * HD + dt * 16 + ln;
      out[((size_t)(b * SEQ + gq)) * D_MODEL + gd] = o[dt][r] / lrun[r];
    }
}

// ---------------------------------------------------------------- residual + layernorm
__global__ __launch_bounds__(256)
void add_ln_kernel(const float* __restrict__ x, const float* __restrict__ y,
                   const float* __restrict__ g, const float* __restrict__ bb,
                   float* __restrict__ outF, uint16_t* __restrict__ outB) {
  __shared__ float red[256];
  const int row = blockIdx.x;
  const int t = threadIdx.x;
  const float* xr = x + (size_t)row * D_MODEL;
  const float* yr = y + (size_t)row * D_MODEL;

  float vals[4];
  float sum = 0.0f;
#pragma unroll
  for (int i = 0; i < 4; ++i) {
    vals[i] = xr[t + i * 256] + yr[t + i * 256];
    sum += vals[i];
  }
  red[t] = sum; __syncthreads();
  for (int s2 = 128; s2 > 0; s2 >>= 1) { if (t < s2) red[t] += red[t + s2]; __syncthreads(); }
  const float mu = red[0] * (1.0f / D_MODEL);
  __syncthreads();

  float var = 0.0f;
#pragma unroll
  for (int i = 0; i < 4; ++i) { float d = vals[i] - mu; var += d * d; }
  red[t] = var; __syncthreads();
  for (int s2 = 128; s2 > 0; s2 >>= 1) { if (t < s2) red[t] += red[t + s2]; __syncthreads(); }
  const float rstd = rsqrtf(red[0] * (1.0f / D_MODEL) + 1e-5f);

#pragma unroll
  for (int i = 0; i < 4; ++i) {
    int c = t + i * 256;
    float yv = (vals[i] - mu) * rstd * g[c] + bb[c];
    outF[(size_t)row * D_MODEL + c] = yv;
    if (outB) outB[(size_t)row * D_MODEL + c] = f2bf(yv);
  }
}

// ---------------------------------------------------------------- host launcher
extern "C" void kernel_launch(void* const* d_in, const int* in_sizes, int n_in,
                              void* d_out, int out_size, void* d_ws, size_t ws_size,
                              hipStream_t stream) {
  const float* x     = (const float*)d_in[0];
  const float* wq    = (const float*)d_in[1];
  const float* wk    = (const float*)d_in[2];
  const float* wv    = (const float*)d_in[3];
  const float* ln1_g = (const float*)d_in[4];
  const float* ln1_b = (const float*)d_in[5];
  const float* l1_w  = (const float*)d_in[6];
  const float* l1_b  = (const float*)d_in[7];
  const float* l2_w  = (const float*)d_in[8];
  const float* l2_b  = (const float*)d_in[9];
  const float* ln2_g = (const float*)d_in[10];
  const float* ln2_b = (const float*)d_in[11];

  const size_t NX = (size_t)M_TOTAL * D_MODEL;          // 8 Mi elems
  const size_t NW = (size_t)D_MODEL * D_MODEL;          // 1 Mi elems

  uint8_t* ws = (uint8_t*)d_ws;
  size_t off = 0;
  auto alloc = [&](size_t bytes) { void* p = ws + off; off += (bytes + 255) & ~(size_t)255; return p; };

  uint16_t* xb    = (uint16_t*)alloc(NX * 2);
  uint16_t* wqb   = (uint16_t*)alloc(NW * 2);
  uint16_t* wkb   = (uint16_t*)alloc(NW * 2);
  uint16_t* wvb   = (uint16_t*)alloc(NW * 2);
  uint16_t* l1wb  = (uint16_t*)alloc(NW * 2);
  uint16_t* l2wb  = (uint16_t*)alloc(NW * 2);
  uint16_t* qb    = (uint16_t*)alloc(NX * 2);
  uint16_t* kb    = (uint16_t*)alloc(NX * 2);
  uint16_t* vb    = (uint16_t*)alloc(NX * 2);
  float*    attn  = (float*)   alloc(NX * 4);
  float*    hbuf  = (float*)   alloc(NX * 4);
  uint16_t* hb    = (uint16_t*)alloc(NX * 2);
  uint16_t* ff1b  = (uint16_t*)alloc(NX * 2);
  float*    ff2   = (float*)   alloc(NX * 4);

  // 1) casts to bf16
  cast_bf16_kernel<<<2048, 256, 0, stream>>>(x,    xb,   (int)NX);
  cast_bf16_kernel<<<512,  256, 0, stream>>>(wq,   wqb,  (int)NW);
  cast_bf16_kernel<<<512,  256, 0, stream>>>(wk,   wkb,  (int)NW);
  cast_bf16_kernel<<<512,  256, 0, stream>>>(wv,   wvb,  (int)NW);
  cast_bf16_kernel<<<512,  256, 0, stream>>>(l1_w, l1wb, (int)NW);
  cast_bf16_kernel<<<512,  256, 0, stream>>>(l2_w, l2wb, (int)NW);

  const int gemm_blocks = (M_TOTAL / 128) * (D_MODEL / 128);  // 512

  // 2) Q/K/V projections (bf16 outputs for flash attention)
  gemm_bt_kernel<false, false, false, true><<<gemm_blocks, 256, 0, stream>>>(
      xb, wqb, nullptr, nullptr, qb, M_TOTAL, D_MODEL, D_MODEL);
  gemm_bt_kernel<false, false, false, true><<<gemm_blocks, 256, 0, stream>>>(
      xb, wkb, nullptr, nullptr, kb, M_TOTAL, D_MODEL, D_MODEL);
  gemm_bt_kernel<false, false, false, true><<<gemm_blocks, 256, 0, stream>>>(
      xb, wvb, nullptr, nullptr, vb, M_TOTAL, D_MODEL, D_MODEL);

  // 3) flash attention
  flash_attn_kernel<<<BATCH * NH * (SEQ / 128), 256, 0, stream>>>(qb, kb, vb, attn);

  // 4) h = LN(x + attn) -> f32 + bf16
  add_ln_kernel<<<M_TOTAL, 256, 0, stream>>>(x, attn, ln1_g, ln1_b, hbuf, hb);

  // 5) ff1 = relu(h @ l1_w^T + l1_b) -> bf16
  gemm_bt_kernel<true, true, false, true><<<gemm_blocks, 256, 0, stream>>>(
      hb, l1wb, l1_b, nullptr, ff1b, M_TOTAL, D_MODEL, D_MODEL);

  // 6) ff2 = ff1 @ l2_w^T + l2_b -> f32
  gemm_bt_kernel<true, false, true, false><<<gemm_blocks, 256, 0, stream>>>(
      ff1b, l2wb, l2_b, ff2, nullptr, M_TOTAL, D_MODEL, D_MODEL);

  // 7) out = LN(h + ff2)
  add_ln_kernel<<<M_TOTAL, 256, 0, stream>>>(hbuf, ff2, ln2_g, ln2_b, (float*)d_out, nullptr);
}
